// SPPNet_81209241633582
// MI455X (gfx1250) — matrix operand order repaired and checked by reading, not compiled, not run on hardware
//
#include <hip/hip_runtime.h>
#include <hip/hip_bf16.h>

// ---------------------------------------------------------------------------
// Types for CDNA5 WMMA (wave32, 16x16x32 bf16 -> f32)
// ---------------------------------------------------------------------------
typedef __bf16 bf16_t;
typedef __attribute__((ext_vector_type(16))) __bf16 v16bf;
typedef __attribute__((ext_vector_type(8)))  __bf16 v8bf;
typedef __attribute__((ext_vector_type(8)))  float  v8f;

__device__ __forceinline__ bf16_t f2bf(float f) {
  union { float f; unsigned u; } v; v.f = f;
  unsigned u = v.u;
  unsigned r = (u + 0x7FFFu + ((u >> 16) & 1u)) >> 16;   // RNE
  unsigned short s = (unsigned short)r;
  bf16_t b; __builtin_memcpy(&b, &s, 2); return b;
}
__device__ __forceinline__ float bf2f(bf16_t b) {
  unsigned short s; __builtin_memcpy(&s, &b, 2);
  union { unsigned u; float f; } v; v.u = ((unsigned)s) << 16; return v.f;
}

// A-fragment (16x32 bf16): lane half cb = (lane>>4)*8 already applied by caller.
// elements 0..7 = p[0..7] (K=cb..cb+7), elements 8..15 = p[16..23] (K=cb+16..)
__device__ __forceinline__ v16bf load_a_frag(const bf16_t* p) {
  v8bf lo = *(const v8bf*)(p);
  v8bf hi = *(const v8bf*)(p + 16);
  v16bf r;
#pragma unroll
  for (int i = 0; i < 8; ++i) { r[i] = lo[i]; r[8 + i] = hi[i]; }
  return r;
}
// B-fragment (32x16 bf16, stored as Bt[n][k] row-major): same K order as A;
// caller passes p already offset by (lane>>4)*8; elements 0..7 = p[0..7],
// elements 8..15 = p[16..23].
__device__ __forceinline__ v16bf load_b_frag(const bf16_t* p) {
  v8bf lo = *(const v8bf*)(p);
  v8bf hi = *(const v8bf*)(p + 16);
  v16bf r;
#pragma unroll
  for (int i = 0; i < 8; ++i) { r[i] = lo[i]; r[8 + i] = hi[i]; }
  return r;
}

#define WMMA_BF16(A, B, C) \
  __builtin_amdgcn_wmma_f32_16x16x32_bf16(false, (A), false, (B), (short)0, (C), false, false)

// CDNA5 async copy: global -> LDS, 16B per active lane, tracked by ASYNCcnt.
// VDST VGPR carries the wave-relative LDS byte address (== low 32 bits of a
// generic __shared__ pointer, per the flat->LDS aperture truncation rule).
__device__ __forceinline__ void async_load_b128(const bf16_t* gsrc, bf16_t* ldst) {
  const unsigned lds = (unsigned)(size_t)ldst;
  const unsigned long long ga = (unsigned long long)(size_t)gsrc;
  asm volatile("global_load_async_to_lds_b128 %0, %1, off"
               :: "v"(lds), "v"(ga) : "memory");
}
__device__ __forceinline__ void wait_asynccnt0() {
  asm volatile("s_wait_asynccnt 0x0" ::: "memory");
}

// ---------------------------------------------------------------------------
// Pipeline: x:(8,3,512,512) -> conv1(64)+pool -> feat1 NHWC bf16 (8,256,256,64)
// -> conv2(128)+pool (WMMA) -> feat NHWC bf16 (8,128,128,128)
// -> ROI+SPP -> s bf16 (256,2688) -> FC1 WMMA -> h bf16 (256,256)
// -> FC2 WMMA -> out f32 (256,1000)
// ---------------------------------------------------------------------------

// Kernel 0a: repack conv2 weights (128,64,3,3) f32 -> bf16 [co][tap*64+ci]
__global__ void k0_repack_w2(const float* __restrict__ w2, bf16_t* __restrict__ w2r) {
  int i = blockIdx.x * 256 + threadIdx.x;
  if (i >= 128 * 576) return;
  int co = i / 576, k = i % 576;
  int tap = k >> 6, ci = k & 63;                 // k = tap*64 + ci
  w2r[i] = f2bf(w2[(co * 64 + ci) * 9 + tap]);   // w2[co][ci][kh][kw], tap=kh*3+kw
}

// Kernel 0b: transpose fc1_w (2688,256) f32 -> bf16 w1t[256][2688]
__global__ void k0_transpose_fc1(const float* __restrict__ fc1_w, bf16_t* __restrict__ w1t) {
  int i = blockIdx.x * 256 + threadIdx.x;
  if (i >= 256 * 2688) return;
  int n = i / 2688, k = i % 2688;
  w1t[i] = f2bf(fc1_w[k * 256 + n]);
}

// Kernel 0c: transpose fc2_w (256,1000) f32 -> bf16 w2t[1008][256] (zero-pad)
__global__ void k0_transpose_fc2(const float* __restrict__ fc2_w, bf16_t* __restrict__ w2t) {
  int i = blockIdx.x * 256 + threadIdx.x;
  if (i >= 1008 * 256) return;
  int n = i / 256, k = i % 256;
  w2t[i] = (n < 1000) ? f2bf(fc2_w[k * 1000 + n]) : f2bf(0.0f);
}

// ---------------------------------------------------------------------------
// Kernel 1: conv1 (3->64, pad 1) + bias + relu + 2x2 maxpool -> feat1 NHWC bf16
// One block = one (b, co, oh2) pooled row; threadIdx.x = ow2 (coalesced in w,
// wave-uniform weights -> scalar loads).
// ---------------------------------------------------------------------------
__global__ void k1_conv1_pool(const float* __restrict__ x, const float* __restrict__ w1,
                              const float* __restrict__ b1, bf16_t* __restrict__ feat1) {
  const int ow2 = threadIdx.x;
  int idx = blockIdx.x;
  const int oh2 = idx & 255; idx >>= 8;
  const int co  = idx & 63;  idx >>= 6;
  const int b   = idx;

  float wv[27];
#pragma unroll
  for (int i = 0; i < 27; ++i) wv[i] = w1[co * 27 + i];   // [co][ci][kh][kw]
  const float bias = b1[co];

  float patch[3][4][4];
#pragma unroll
  for (int ci = 0; ci < 3; ++ci) {
#pragma unroll
    for (int dy = 0; dy < 4; ++dy) {
      const int ih = 2 * oh2 - 1 + dy;
#pragma unroll
      for (int dx = 0; dx < 4; ++dx) {
        const int iw = 2 * ow2 - 1 + dx;
        float v = 0.0f;
        if (ih >= 0 && ih < 512 && iw >= 0 && iw < 512)
          v = x[(((size_t)b * 3 + ci) * 512 + ih) * 512 + iw];
        patch[ci][dy][dx] = v;
      }
    }
  }

  float best = 0.0f;                      // relu floor folded into the max
#pragma unroll
  for (int sy = 0; sy < 2; ++sy) {
#pragma unroll
    for (int sx = 0; sx < 2; ++sx) {
      float sum = bias;
#pragma unroll
      for (int ci = 0; ci < 3; ++ci)
#pragma unroll
        for (int kh = 0; kh < 3; ++kh)
#pragma unroll
          for (int kw = 0; kw < 3; ++kw)
            sum += patch[ci][sy + kh][sx + kw] * wv[ci * 9 + kh * 3 + kw];
      best = fmaxf(best, sum);
    }
  }
  feat1[(((size_t)b * 256 + oh2) * 256 + ow2) * 64 + co] = f2bf(best);
}

// ---------------------------------------------------------------------------
// Kernel 2: conv2 (64->128, pad 1) + bias + relu + 2x2 maxpool via WMMA bf16.
// Block: 256 threads (8 waves) covers a 16x16 pre-pool tile of one image.
// LDS: 18x18 halo patch, 64 channels padded to 72 (bank-conflict-free).
// Halo staged with GLOBAL_LOAD_ASYNC_TO_LDS_B128 (ASYNCcnt); border pixels
// zero-filled with ds stores. Wave w owns N-tile co = w*16..w*16+15 and
// preloads all 18 B fragments in registers (weight row prefetched to cache).
// K order = (tap, ci): 9 taps x 2 ci-halves of 32 = 18 WMMA steps per M-tile.
// ---------------------------------------------------------------------------
__global__ void __launch_bounds__(256)
k2_conv2_wmma(const bf16_t* __restrict__ feat1, const bf16_t* __restrict__ w2r,
              const float* __restrict__ b2, bf16_t* __restrict__ feat) {
  __shared__ __align__(16) bf16_t patch[18 * 18 * 72];

  const int tid = threadIdx.x;
  const int tx = blockIdx.x, ty = blockIdx.y, b = blockIdx.z;

  const int lane = tid & 31;
  const int wave = tid >> 5;               // N-tile (8 waves -> 128 channels)
  const int mrow = lane & 15;
  const int cb   = (lane >> 4) * 8;        // A-fragment half
  const int kb   = (lane >> 4) * 8;        // B-fragment half
  const int co   = wave * 16 + mrow;
  const bf16_t* wrow = w2r + (size_t)co * 576;
  __builtin_prefetch(wrow, 0, 0);          // global_prefetch_b8 (weight row)

  // ---- stage halo patch (zero-padded borders), 128B per pixel ----
  for (int p = tid; p < 18 * 18; p += 256) {
    const int ph = p / 18, pw = p % 18;
    const int ih = ty * 16 - 1 + ph;
    const int iw = tx * 16 - 1 + pw;
    bf16_t* dst = &patch[p * 72];
    if (ih >= 0 && ih < 256 && iw >= 0 && iw < 256) {
      const bf16_t* src = feat1 + (((size_t)b * 256 + ih) * 256 + iw) * 64;
#pragma unroll
      for (int v = 0; v < 8; ++v)
        async_load_b128(src + v * 8, dst + v * 8);   // 8 x 16B -> LDS, ASYNCcnt
    } else {
      unsigned* d32 = (unsigned*)dst;
#pragma unroll
      for (int v = 0; v < 32; ++v) d32[v] = 0u;      // ds stores (DScnt)
    }
  }
  wait_asynccnt0();                        // drain this wave's async copies
  __syncthreads();                         // + barrier drains ds stores

  const float bias = b2[co];

  // Preload all 18 B fragments for this wave's 16 output channels.
  v16bf bfrag[18];
#pragma unroll
  for (int ks = 0; ks < 18; ++ks)
    bfrag[ks] = load_b_frag(wrow + ks * 32 + kb);

  const int phbase = ty * 8;
  const int pwbase = tx * 8 + (lane >> 4) * 4;

  for (int mt = 0; mt < 16; mt += 2) {     // pre-pool row pair -> one pooled row
    v8f acc0 = {}; v8f acc1 = {};
#pragma unroll
    for (int ks = 0; ks < 18; ++ks) {
      const int tap = ks >> 1;
      const int kh = tap / 3, kw = tap - kh * 3;
      const int ci0 = (ks & 1) * 32;
      const bf16_t* a0 = &patch[(((mt + 0) + kh) * 18 + (mrow + kw)) * 72 + ci0 + cb];
      acc0 = WMMA_BF16(load_a_frag(a0), bfrag[ks], acc0);
      const bf16_t* a1 = a0 + 18 * 72;     // row mt+1
      acc1 = WMMA_BF16(load_a_frag(a1), bfrag[ks], acc1);
    }
    // relu + 2x2 maxpool directly from accumulator layout:
    // VGPR r holds pixel mw=r (lanes<16) / mw=8+r (lanes>=16).
    const int ph = phbase + (mt >> 1);
    bf16_t* orow = feat + (((size_t)b * 128 + ph) * 128) * 128 + co;
#pragma unroll
    for (int p = 0; p < 4; ++p) {
      float m = fmaxf(fmaxf(acc0[2 * p], acc0[2 * p + 1]),
                      fmaxf(acc1[2 * p], acc1[2 * p + 1]));
      m = fmaxf(m + bias, 0.0f);
      orow[(size_t)(pwbase + p) * 128] = f2bf(m);
    }
  }
}

// ---------------------------------------------------------------------------
// Kernel 3: ROI crop + SPP max pooling -> s bf16 [256][2688]
// s layout: [0,128)=p1, [128,640)=p2 (c*4+cell), [640,2688)=p4 (c*16+cell)
// ---------------------------------------------------------------------------
__global__ void k3_roi_spp(const bf16_t* __restrict__ feat, const int* __restrict__ rois,
                           bf16_t* __restrict__ s) {
  __shared__ float cellmax[128 * 16];
  const int n = blockIdx.x;
  const int t = threadIdx.x;
  const int b = rois[n * 5 + 0];
  int x0 = (int)floorf((float)rois[n * 5 + 1] * 0.25f + 0.5f);
  int y0 = (int)floorf((float)rois[n * 5 + 2] * 0.25f + 0.5f);
  x0 = min(max(x0, 0), 128 - 32);
  y0 = min(max(y0, 0), 128 - 32);

  for (int q = t; q < 2048; q += 256) {       // 128 ch x 16 cells of 8x8
    const int c = q >> 4, cell = q & 15;
    const int py = cell >> 2, px = cell & 3;
    const bf16_t* base =
        feat + (((size_t)b * 128 + (y0 + py * 8)) * 128 + (x0 + px * 8)) * 128 + c;
    float m = -3.4e38f;
    for (int dy = 0; dy < 8; ++dy)
#pragma unroll
      for (int dx = 0; dx < 8; ++dx)
        m = fmaxf(m, bf2f(base[(dy * 128 + dx) * 128]));
    cellmax[q] = m;
    s[(size_t)n * 2688 + 640 + q] = f2bf(m);  // p=4 block
  }
  __syncthreads();
  for (int q = t; q < 512; q += 256) {        // p=2
    const int c = q >> 2, cell2 = q & 3;
    const int py2 = cell2 >> 1, px2 = cell2 & 1;
    const float* cm = &cellmax[c * 16];
    float m = -3.4e38f;
#pragma unroll
    for (int dy = 0; dy < 2; ++dy)
#pragma unroll
      for (int dx = 0; dx < 2; ++dx)
        m = fmaxf(m, cm[(py2 * 2 + dy) * 4 + (px2 * 2 + dx)]);
    s[(size_t)n * 2688 + 128 + q] = f2bf(m);
  }
  for (int c = t; c < 128; c += 256) {        // p=1
    const float* cm = &cellmax[c * 16];
    float m = cm[0];
#pragma unroll
    for (int i = 1; i < 16; ++i) m = fmaxf(m, cm[i]);
    s[(size_t)n * 2688 + c] = f2bf(m);
  }
}

// ---------------------------------------------------------------------------
// Kernel 4: FC1 h = relu(s @ fc1_w + b): M=256, N=256, K=2688 (84 WMMA steps)
// One wave per 16x16 output tile; 32 blocks x 8 waves = 256 tiles.
// ---------------------------------------------------------------------------
__global__ void __launch_bounds__(256)
k4_fc1(const bf16_t* __restrict__ s, const bf16_t* __restrict__ w1t,
       const float* __restrict__ fc1_b, bf16_t* __restrict__ h) {
  const int lane = threadIdx.x & 31;
  const int wave = threadIdx.x >> 5;
  const int gw = blockIdx.x * 8 + wave;
  const int mt = gw >> 4, nt = gw & 15;
  const int mrow = lane & 15;
  const int cb = (lane >> 4) * 8;
  const int kb = (lane >> 4) * 8;
  const bf16_t* arow = s   + (size_t)(mt * 16 + mrow) * 2688;
  const bf16_t* brow = w1t + (size_t)(nt * 16 + mrow) * 2688;
  __builtin_prefetch(arow, 0, 0);
  __builtin_prefetch(brow, 0, 0);
  v8f acc = {};
#pragma unroll 4
  for (int ks = 0; ks < 84; ++ks) {
    acc = WMMA_BF16(load_a_frag(arow + ks * 32 + cb),
                    load_b_frag(brow + ks * 32 + kb), acc);
  }
  const int ncol  = nt * 16 + mrow;
  const float bias = fc1_b[ncol];
  const int mbase = mt * 16 + (lane >> 4) * 8;
#pragma unroll
  for (int r = 0; r < 8; ++r)
    h[(size_t)(mbase + r) * 256 + ncol] = f2bf(fmaxf(acc[r] + bias, 0.0f));
}

// ---------------------------------------------------------------------------
// Kernel 5: FC2 out = h @ fc2_w + b: M=256, N=1008 (pad of 1000), K=256.
// 126 blocks x 8 waves = 1008 tiles; stores guarded to n < 1000, f32 out.
// ---------------------------------------------------------------------------
__global__ void __launch_bounds__(256)
k5_fc2(const bf16_t* __restrict__ h, const bf16_t* __restrict__ w2t,
       const float* __restrict__ fc2_b, float* __restrict__ out) {
  const int lane = threadIdx.x & 31;
  const int wave = threadIdx.x >> 5;
  const int gw = blockIdx.x * 8 + wave;
  const int mt = gw / 63, nt = gw % 63;
  const int mrow = lane & 15;
  const int cb = (lane >> 4) * 8;
  const int kb = (lane >> 4) * 8;
  const bf16_t* arow = h   + (size_t)(mt * 16 + mrow) * 256;
  const bf16_t* brow = w2t + (size_t)(nt * 16 + mrow) * 256;
  v8f acc = {};
#pragma unroll
  for (int ks = 0; ks < 8; ++ks) {
    acc = WMMA_BF16(load_a_frag(arow + ks * 32 + cb),
                    load_b_frag(brow + ks * 32 + kb), acc);
  }
  const int ncol = nt * 16 + mrow;
  if (ncol < 1000) {                        // divergence only after all WMMAs
    const float bias = fc2_b[ncol];
    const int mbase = mt * 16 + (lane >> 4) * 8;
#pragma unroll
    for (int r = 0; r < 8; ++r)
      out[(size_t)(mbase + r) * 1000 + ncol] = acc[r] + bias;
  }
}

// ---------------------------------------------------------------------------
// Launch: workspace layout (bytes, all 256-aligned); total ~100 MB.
// ---------------------------------------------------------------------------
extern "C" void kernel_launch(void* const* d_in, const int* in_sizes, int n_in,
                              void* d_out, int out_size, void* d_ws, size_t ws_size,
                              hipStream_t stream) {
  (void)in_sizes; (void)n_in; (void)out_size; (void)ws_size;
  const float* x     = (const float*)d_in[0];
  const int*   rois  = (const int*)  d_in[1];
  const float* w1    = (const float*)d_in[2];
  const float* b1    = (const float*)d_in[3];
  const float* w2    = (const float*)d_in[4];
  const float* b2    = (const float*)d_in[5];
  const float* fc1_w = (const float*)d_in[6];
  const float* fc1_b = (const float*)d_in[7];
  const float* fc2_w = (const float*)d_in[8];
  const float* fc2_b = (const float*)d_in[9];
  float* out = (float*)d_out;

  char* ws = (char*)d_ws;
  bf16_t* feat1 = (bf16_t*)(ws + 0);           //  8*256*256*64 bf16 = 64 MB
  bf16_t* feat  = (bf16_t*)(ws + 67108864);    //  8*128*128*128 bf16 = 32 MB
  bf16_t* w2r   = (bf16_t*)(ws + 100663296);   //  128*576 bf16
  bf16_t* w1t   = (bf16_t*)(ws + 100810752);   //  256*2688 bf16
  bf16_t* w2t   = (bf16_t*)(ws + 102187008);   //  1008*256 bf16
  bf16_t* sbuf  = (bf16_t*)(ws + 102703104);   //  256*2688 bf16
  bf16_t* hbuf  = (bf16_t*)(ws + 104079360);   //  256*256 bf16

  k0_repack_w2    <<<(128 * 576  + 255) / 256, 256, 0, stream>>>(w2, w2r);
  k0_transpose_fc1<<<(256 * 2688 + 255) / 256, 256, 0, stream>>>(fc1_w, w1t);
  k0_transpose_fc2<<<(1008 * 256 + 255) / 256, 256, 0, stream>>>(fc2_w, w2t);
  k1_conv1_pool   <<<8 * 64 * 256, 256, 0, stream>>>(x, w1, b1, feat1);
  k2_conv2_wmma   <<<dim3(16, 16, 8), 256, 0, stream>>>(feat1, w2r, b2, feat);
  k3_roi_spp      <<<256, 256, 0, stream>>>(feat, rois, sbuf);
  k4_fc1          <<<32, 256, 0, stream>>>(sbuf, w1t, fc1_b, hbuf);
  k5_fc2          <<<126, 256, 0, stream>>>(hbuf, w2t, fc2_b, out);
}
